// LocalBlock3D_28647431864485
// MI455X (gfx1250) — matrix operand order repaired and checked by reading, not compiled, not run on hardware
//
#include <hip/hip_runtime.h>
#include <hip/hip_bf16.h>
#include <math.h>

typedef __attribute__((ext_vector_type(16))) __bf16 v16bf;
typedef __attribute__((ext_vector_type(8)))  __bf16 v8bf;
typedef __attribute__((ext_vector_type(8)))  float  v8f;

#define NTOK 4096      // 16*16*16
#define CW   128       // width
#define HID  512       // mlp hidden
#define EPSF 1e-6f

// attention LDS tiling
#define TS        4                    // voxel tile edge
#define HALO      8                    // TS + 2*radius
#define NHALO     (HALO * HALO * HALO) // 512 halo voxels
#define PADC      136                  // padded row stride in bf16 elems (272B, breaks 256B bank period)
#define ATTN_LDS  (2u * NHALO * PADC * 2u)  // K + V, bytes (278528 <= 320KB)

// ---------------------------------------------------------------------------
// CDNA5 async global->LDS copy (16B per lane) + completion wait
// ---------------------------------------------------------------------------
__device__ __forceinline__ void async_copy_b128(unsigned lds_off, const void* gaddr) {
    asm volatile("global_load_async_to_lds_b128 %0, %1, off"
                 :: "v"(lds_off), "v"(gaddr) : "memory");
}
__device__ __forceinline__ void wait_async0() {
    asm volatile("s_wait_asynccnt 0" ::: "memory");
}

// ---------------------------------------------------------------------------
// A-fragment loader: row-major bf16 A[M,K]; lane layout per CDNA5 ISA
// ---------------------------------------------------------------------------
__device__ __forceinline__ v16bf load_a_frag(const __bf16* arow, int kt) {
    v8bf a0 = *(const v8bf*)(arow + kt * 32);
    v8bf a1 = *(const v8bf*)(arow + kt * 32 + 16);
    v16bf a;
    #pragma unroll
    for (int i = 0; i < 8; ++i) { a[i] = a0[i]; a[i + 8] = a1[i]; }
    return a;
}

// ---------------------------------------------------------------------------
// RMSNorm (over last dim 128) + cast to bf16  (one thread per row)
// ---------------------------------------------------------------------------
__global__ void k_rms_cast(const float* __restrict__ x, const float* __restrict__ w,
                           __bf16* __restrict__ out, int nrows) {
    int row = blockIdx.x * blockDim.x + threadIdx.x;
    if (row >= nrows) return;
    const float* xr = x + (size_t)row * CW;
    float ss = 0.f;
    #pragma unroll
    for (int i = 0; i < CW; i += 4) {
        float4 v = *(const float4*)(xr + i);
        ss += v.x*v.x + v.y*v.y + v.z*v.z + v.w*v.w;
    }
    float r = rsqrtf(ss * (1.f / CW) + EPSF);
    __bf16* orow = out + (size_t)row * CW;
    #pragma unroll
    for (int i = 0; i < CW; i += 4) {
        float4 v = *(const float4*)(xr + i);
        orow[i + 0] = (__bf16)(v.x * r * w[i + 0]);
        orow[i + 1] = (__bf16)(v.y * r * w[i + 1]);
        orow[i + 2] = (__bf16)(v.z * r * w[i + 2]);
        orow[i + 3] = (__bf16)(v.w * r * w[i + 3]);
    }
}

// ---------------------------------------------------------------------------
// Pack row-major f32 weight [K,N] into WMMA B-operand bf16 lane layout.
// ---------------------------------------------------------------------------
__global__ void k_pack_b(const float* __restrict__ W, __bf16* __restrict__ Bp,
                         int K, int N) {
    int idx = blockIdx.x * blockDim.x + threadIdx.x;
    if (idx >= K * N) return;
    int Ktiles = K >> 5;
    int i    = idx & 15;
    int lane = (idx >> 4) & 31;
    int blk  = idx >> 9;
    int kt   = blk % Ktiles;
    int nt   = blk / Ktiles;
    int col  = nt * 16 + (lane & 15);
    int krow = kt * 32 + ((i >> 3) << 4) + ((lane >> 4) << 3) + (i & 7);
    Bp[idx] = (__bf16)W[(size_t)krow * N + col];
}

// ---------------------------------------------------------------------------
// Fused QKV GEMM. Q out f32; K,V out bf16 (feed the LDS attention tile).
// blockDim=128 (4 waves), grid=(M/16, N/64).
// ---------------------------------------------------------------------------
__global__ void k_gemm_qkv(const __bf16* __restrict__ A,
                           const __bf16* __restrict__ Bq,
                           const __bf16* __restrict__ Bk,
                           const __bf16* __restrict__ Bv,
                           float* __restrict__ Q, __bf16* __restrict__ Ko,
                           __bf16* __restrict__ V) {
    constexpr int Ktiles = CW / 32;
    int lane = threadIdx.x & 31;
    int wv   = threadIdx.x >> 5;
    int mt   = blockIdx.x;
    int nt   = blockIdx.y * 4 + wv;
    int half = lane >> 4, l16 = lane & 15;
    const __bf16* arow = A + (size_t)(mt * 16 + l16) * CW + half * 8;
    size_t boff = ((size_t)nt * Ktiles * 32 + lane) * 16;

    v8f aq = {}, ak = {}, av = {};
    #pragma unroll
    for (int kt = 0; kt < Ktiles; ++kt) {
        v16bf a = load_a_frag(arow, kt);
        v16bf bq = *(const v16bf*)(Bq + boff + (size_t)kt * 512);
        v16bf bk = *(const v16bf*)(Bk + boff + (size_t)kt * 512);
        v16bf bv = *(const v16bf*)(Bv + boff + (size_t)kt * 512);
        aq = __builtin_amdgcn_wmma_f32_16x16x32_bf16(false, a, false, bq, (short)0, aq, false, false);
        ak = __builtin_amdgcn_wmma_f32_16x16x32_bf16(false, a, false, bk, (short)0, ak, false, false);
        av = __builtin_amdgcn_wmma_f32_16x16x32_bf16(false, a, false, bv, (short)0, av, false, false);
    }
    size_t base = (size_t)(mt * 16 + half * 8) * CW + nt * 16 + l16;
    #pragma unroll
    for (int r = 0; r < 8; ++r) {
        Q [base + (size_t)r * CW] = aq[r];
        Ko[base + (size_t)r * CW] = (__bf16)ak[r];
        V [base + (size_t)r * CW] = (__bf16)av[r];
    }
}

// ---------------------------------------------------------------------------
// Plain GEMM (o @ wo): one wave -> 16x64, A fragment reused 4x.
// blockDim=32, grid=(M/16, N/64).
// ---------------------------------------------------------------------------
template <int K>
__global__ void k_gemm_plain(const __bf16* __restrict__ A,
                             const __bf16* __restrict__ Bp,
                             float* __restrict__ C, int N) {
    constexpr int Ktiles = K / 32;
    int lane = threadIdx.x & 31;
    int mt   = blockIdx.x;
    int ntb  = blockIdx.y * 4;
    int half = lane >> 4, l16 = lane & 15;
    const __bf16* arow = A + (size_t)(mt * 16 + l16) * K + half * 8;

    v8f acc[4] = {};
    #pragma unroll
    for (int kt = 0; kt < Ktiles; ++kt) {
        v16bf a = load_a_frag(arow, kt);
        #pragma unroll
        for (int j = 0; j < 4; ++j) {
            const __bf16* bp = Bp + (((size_t)(ntb + j) * Ktiles + kt) * 32 + lane) * 16;
            v16bf b = *(const v16bf*)bp;
            acc[j] = __builtin_amdgcn_wmma_f32_16x16x32_bf16(false, a, false, b, (short)0, acc[j], false, false);
        }
    }
    #pragma unroll
    for (int j = 0; j < 4; ++j) {
        float* crow = C + (size_t)(mt * 16 + half * 8) * N + (ntb + j) * 16 + l16;
        #pragma unroll
        for (int r = 0; r < 8; ++r) crow[(size_t)r * N] = acc[j][r];
    }
}

// ---------------------------------------------------------------------------
// Fused gate/up GEMM + SwiGLU epilogue: m = bf16(silu(h2@gate) * (h2@up)).
// ---------------------------------------------------------------------------
__global__ void k_gemm_gateup(const __bf16* __restrict__ A,
                              const __bf16* __restrict__ Bg,
                              const __bf16* __restrict__ Bu,
                              __bf16* __restrict__ Mo) {
    constexpr int Ktiles = CW / 32;
    int lane = threadIdx.x & 31;
    int wv   = threadIdx.x >> 5;
    int mt   = blockIdx.x;
    int nt   = blockIdx.y * 4 + wv;
    int half = lane >> 4, l16 = lane & 15;
    const __bf16* arow = A + (size_t)(mt * 16 + l16) * CW + half * 8;
    size_t boff = ((size_t)nt * Ktiles * 32 + lane) * 16;

    v8f ag = {}, au = {};
    #pragma unroll
    for (int kt = 0; kt < Ktiles; ++kt) {
        v16bf a = load_a_frag(arow, kt);
        v16bf bg = *(const v16bf*)(Bg + boff + (size_t)kt * 512);
        v16bf bu = *(const v16bf*)(Bu + boff + (size_t)kt * 512);
        ag = __builtin_amdgcn_wmma_f32_16x16x32_bf16(false, a, false, bg, (short)0, ag, false, false);
        au = __builtin_amdgcn_wmma_f32_16x16x32_bf16(false, a, false, bu, (short)0, au, false, false);
    }
    __bf16* mrow = Mo + (size_t)(mt * 16 + half * 8) * HID + nt * 16 + l16;
    #pragma unroll
    for (int r = 0; r < 8; ++r) {
        float g = ag[r];
        float s = g / (1.f + __expf(-g));
        mrow[(size_t)r * HID] = (__bf16)(s * au[r]);
    }
}

// ---------------------------------------------------------------------------
// Down GEMM (K=512) fused with final residual: out = x1 + m @ down.
// blockDim=32, grid=(M/16, CW/64).
// ---------------------------------------------------------------------------
__global__ void k_gemm_down_add(const __bf16* __restrict__ A,
                                const __bf16* __restrict__ Bp,
                                const float* __restrict__ x1,
                                float* __restrict__ out) {
    constexpr int Ktiles = HID / 32;
    int lane = threadIdx.x & 31;
    int mt   = blockIdx.x;
    int ntb  = blockIdx.y * 4;
    int half = lane >> 4, l16 = lane & 15;
    const __bf16* arow = A + (size_t)(mt * 16 + l16) * HID + half * 8;

    v8f acc[4] = {};
    #pragma unroll
    for (int kt = 0; kt < Ktiles; ++kt) {
        v16bf a = load_a_frag(arow, kt);
        #pragma unroll
        for (int j = 0; j < 4; ++j) {
            const __bf16* bp = Bp + (((size_t)(ntb + j) * Ktiles + kt) * 32 + lane) * 16;
            v16bf b = *(const v16bf*)bp;
            acc[j] = __builtin_amdgcn_wmma_f32_16x16x32_bf16(false, a, false, b, (short)0, acc[j], false, false);
        }
    }
    #pragma unroll
    for (int j = 0; j < 4; ++j) {
        size_t base = (size_t)(mt * 16 + half * 8) * CW + (ntb + j) * 16 + l16;
        #pragma unroll
        for (int r = 0; r < 8; ++r)
            out[base + (size_t)r * CW] = x1[base + (size_t)r * CW] + acc[j][r];
    }
}

// ---------------------------------------------------------------------------
// 5x5x5 local-window attention with LDS halo tile.
// Block = 4x4x4 voxel tile, 256 threads = (64 voxels x 4 heads).
// K,V halo (8x8x8 voxels, bf16, padded stride) staged via async global->LDS.
// ---------------------------------------------------------------------------
__global__ void k_attn_lds(const float* __restrict__ Q,
                           const __bf16* __restrict__ Kb,
                           const __bf16* __restrict__ Vb,
                           __bf16* __restrict__ O) {
    extern __shared__ __bf16 smem[];
    __bf16* sK = smem;
    __bf16* sV = smem + NHALO * PADC;

    int bt = blockIdx.x;                 // 64 tiles
    int tx = (bt & 3) * TS;
    int ty = ((bt >> 2) & 3) * TS;
    int tz = (bt >> 4) * TS;
    int ox = tx - 2, oy = ty - 2, oz = tz - 2;

    // cooperative async halo fill: 512 voxels / 256 threads = 2 rows each
    for (int it = threadIdx.x; it < NHALO; it += 256) {
        int lx = it & 7, ly = (it >> 3) & 7, lz = it >> 6;
        int gx = ox + lx, gy = oy + ly, gz = oz + lz;
        if (((unsigned)gx < 16u) && ((unsigned)gy < 16u) && ((unsigned)gz < 16u)) {
            int g = ((gz * 16) + gy) * 16 + gx;
            const __bf16* kg = Kb + (size_t)g * CW;
            const __bf16* vg = Vb + (size_t)g * CW;
            unsigned kl = (unsigned)(uintptr_t)(sK + (size_t)it * PADC);
            unsigned vl = (unsigned)(uintptr_t)(sV + (size_t)it * PADC);
            #pragma unroll
            for (int c = 0; c < CW; c += 8) {      // 16B per async op
                async_copy_b128(kl + c * 2, kg + c);
                async_copy_b128(vl + c * 2, vg + c);
            }
        }
    }
    wait_async0();
    __syncthreads();

    int head = threadIdx.x & 3;
    int vox  = threadIdx.x >> 2;               // 0..63
    int lx0 = vox & 3, ly0 = (vox >> 2) & 3, lz0 = vox >> 4;
    int gx0 = tx + lx0, gy0 = ty + ly0, gz0 = tz + lz0;
    int n = ((gz0 * 16) + gy0) * 16 + gx0;

    const float* qv = Q + (size_t)n * CW + head * 32;
    float q[32];
    #pragma unroll
    for (int e = 0; e < 32; e += 4) {
        float4 t = *(const float4*)(qv + e);
        q[e] = t.x; q[e+1] = t.y; q[e+2] = t.z; q[e+3] = t.w;
    }
    float mmax = -3.0e38f, l = 0.f;
    float acc[32];
    #pragma unroll
    for (int e = 0; e < 32; ++e) acc[e] = 0.f;
    const float scale = 0.17677669529663687f;  // 1/sqrt(32)

    for (int dz = -2; dz <= 2; ++dz) {
        int z = gz0 + dz; if ((unsigned)z >= 16u) continue;
        for (int dy = -2; dy <= 2; ++dy) {
            int y = gy0 + dy; if ((unsigned)y >= 16u) continue;
            for (int dx = -2; dx <= 2; ++dx) {
                int x = gx0 + dx; if ((unsigned)x >= 16u) continue;
                int hl = (((lz0 + 2 + dz) * HALO) + (ly0 + 2 + dy)) * HALO + (lx0 + 2 + dx);
                const __bf16* kv = sK + (size_t)hl * PADC + head * 32;
                float s = 0.f;
                #pragma unroll
                for (int e = 0; e < 32; e += 8) {
                    v8bf kk = *(const v8bf*)(kv + e);
                    #pragma unroll
                    for (int i = 0; i < 8; ++i) s += q[e + i] * (float)kk[i];
                }
                s *= scale;
                float mn = fmaxf(mmax, s);
                float co = __expf(mmax - mn);
                float p  = __expf(s - mn);
                l = l * co + p;
                const __bf16* vvp = sV + (size_t)hl * PADC + head * 32;
                #pragma unroll
                for (int e = 0; e < 32; e += 8) {
                    v8bf vv = *(const v8bf*)(vvp + e);
                    #pragma unroll
                    for (int i = 0; i < 8; ++i) acc[e + i] = acc[e + i] * co + p * (float)vv[i];
                }
                mmax = mn;
            }
        }
    }
    float rl = 1.f / l;
    __bf16* orow = O + (size_t)n * CW + head * 32;
    #pragma unroll
    for (int e = 0; e < 32; ++e) orow[e] = (__bf16)(acc[e] * rl);
}

// ---------------------------------------------------------------------------
// x1 = x + rmsnorm(t, wa);  h2 = bf16(rmsnorm(x1, w2)).  One thread per row.
// ---------------------------------------------------------------------------
__global__ void k_res_rms(const float* __restrict__ x, const float* __restrict__ t,
                          const float* __restrict__ wa, const float* __restrict__ w2,
                          float* __restrict__ x1, __bf16* __restrict__ h2, int nrows) {
    int row = blockIdx.x * blockDim.x + threadIdx.x;
    if (row >= nrows) return;
    const float* tr = t + (size_t)row * CW;
    const float* xr = x + (size_t)row * CW;
    float ss = 0.f;
    #pragma unroll
    for (int i = 0; i < CW; i += 4) {
        float4 v = *(const float4*)(tr + i);
        ss += v.x*v.x + v.y*v.y + v.z*v.z + v.w*v.w;
    }
    float r = rsqrtf(ss * (1.f / CW) + EPSF);
    float* x1r = x1 + (size_t)row * CW;
    float ss2 = 0.f;
    #pragma unroll
    for (int i = 0; i < CW; i += 4) {
        float4 tv = *(const float4*)(tr + i);
        float4 xv = *(const float4*)(xr + i);
        float4 o;
        o.x = xv.x + tv.x * r * wa[i + 0];
        o.y = xv.y + tv.y * r * wa[i + 1];
        o.z = xv.z + tv.z * r * wa[i + 2];
        o.w = xv.w + tv.w * r * wa[i + 3];
        *(float4*)(x1r + i) = o;
        ss2 += o.x*o.x + o.y*o.y + o.z*o.z + o.w*o.w;
    }
    float r2 = rsqrtf(ss2 * (1.f / CW) + EPSF);
    __bf16* h2r = h2 + (size_t)row * CW;
    #pragma unroll
    for (int i = 0; i < CW; i += 4) {
        float4 o = *(const float4*)(x1r + i);
        h2r[i + 0] = (__bf16)(o.x * r2 * w2[i + 0]);
        h2r[i + 1] = (__bf16)(o.y * r2 * w2[i + 1]);
        h2r[i + 2] = (__bf16)(o.z * r2 * w2[i + 2]);
        h2r[i + 3] = (__bf16)(o.w * r2 * w2[i + 3]);
    }
}

// ---------------------------------------------------------------------------
extern "C" void kernel_launch(void* const* d_in, const int* in_sizes, int n_in,
                              void* d_out, int out_size, void* d_ws, size_t ws_size,
                              hipStream_t stream) {
    const float* x       = (const float*)d_in[0];
    const float* norm1_w = (const float*)d_in[1];
    const float* attn_nw = (const float*)d_in[2];
    const float* norm2_w = (const float*)d_in[3];
    const float* wq      = (const float*)d_in[4];
    const float* wk      = (const float*)d_in[5];
    const float* wv      = (const float*)d_in[6];
    const float* wo      = (const float*)d_in[7];
    const float* gate_w  = (const float*)d_in[8];
    const float* up_w    = (const float*)d_in[9];
    const float* down_w  = (const float*)d_in[10];
    float* out = (float*)d_out;

    char* ws = (char*)d_ws;
    size_t off = 0;
    auto alloc = [&](size_t bytes) -> char* {
        char* p = ws + off;
        off = (off + bytes + 255) & ~(size_t)255;
        return p;
    };
    __bf16* h_b    = (__bf16*)alloc((size_t)NTOK * CW * 2);
    float*  qb     = (float*) alloc((size_t)NTOK * CW * 4);
    __bf16* kb     = (__bf16*)alloc((size_t)NTOK * CW * 2);
    __bf16* vb     = (__bf16*)alloc((size_t)NTOK * CW * 2);
    __bf16* o_b    = (__bf16*)alloc((size_t)NTOK * CW * 2);
    float*  t_f    = (float*) alloc((size_t)NTOK * CW * 4);
    float*  x1_f   = (float*) alloc((size_t)NTOK * CW * 4);
    __bf16* h2_b   = (__bf16*)alloc((size_t)NTOK * CW * 2);
    __bf16* m_b    = (__bf16*)alloc((size_t)NTOK * HID * 2);
    __bf16* wq_p   = (__bf16*)alloc((size_t)CW * CW * 2);
    __bf16* wk_p   = (__bf16*)alloc((size_t)CW * CW * 2);
    __bf16* wv_p   = (__bf16*)alloc((size_t)CW * CW * 2);
    __bf16* wo_p   = (__bf16*)alloc((size_t)CW * CW * 2);
    __bf16* gate_p = (__bf16*)alloc((size_t)CW * HID * 2);
    __bf16* up_p   = (__bf16*)alloc((size_t)CW * HID * 2);
    __bf16* down_p = (__bf16*)alloc((size_t)HID * CW * 2);
    (void)ws_size; (void)in_sizes; (void)n_in; (void)out_size;

    // weight packing (bf16, WMMA B layout)
    k_pack_b<<<(CW * CW + 255) / 256, 256, 0, stream>>>(wq, wq_p, CW, CW);
    k_pack_b<<<(CW * CW + 255) / 256, 256, 0, stream>>>(wk, wk_p, CW, CW);
    k_pack_b<<<(CW * CW + 255) / 256, 256, 0, stream>>>(wv, wv_p, CW, CW);
    k_pack_b<<<(CW * CW + 255) / 256, 256, 0, stream>>>(wo, wo_p, CW, CW);
    k_pack_b<<<(CW * HID + 255) / 256, 256, 0, stream>>>(gate_w, gate_p, CW, HID);
    k_pack_b<<<(CW * HID + 255) / 256, 256, 0, stream>>>(up_w, up_p, CW, HID);
    k_pack_b<<<(HID * CW + 255) / 256, 256, 0, stream>>>(down_w, down_p, HID, CW);

    // h = rmsnorm(x) -> bf16
    k_rms_cast<<<NTOK / 256, 256, 0, stream>>>(x, norm1_w, h_b, NTOK);

    // q (f32), k,v (bf16) = h @ wq/wk/wv (fused, shared A fragments)
    k_gemm_qkv<<<dim3(NTOK / 16, CW / 64), 128, 0, stream>>>(h_b, wq_p, wk_p, wv_p, qb, kb, vb);

    // local-window attention with LDS halo tiles -> o (bf16)
    k_attn_lds<<<64, 256, ATTN_LDS, stream>>>(qb, kb, vb, o_b);

    // t = o @ wo
    k_gemm_plain<CW><<<dim3(NTOK / 16, CW / 64), 32, 0, stream>>>(o_b, wo_p, t_f, CW);

    // x1 = x + rmsnorm(t); h2 = bf16(rmsnorm(x1))
    k_res_rms<<<NTOK / 256, 256, 0, stream>>>(x, t_f, attn_nw, norm2_w, x1_f, h2_b, NTOK);

    // m = bf16(silu(h2@gate) * (h2@up))   (fused SwiGLU epilogue)
    k_gemm_gateup<<<dim3(NTOK / 16, HID / 64), 128, 0, stream>>>(h2_b, gate_p, up_p, m_b);

    // out = x1 + m @ down   (fused residual epilogue, K=512)
    k_gemm_down_add<<<dim3(NTOK / 16, CW / 64), 32, 0, stream>>>(m_b, down_p, x1_f, out);
}
